// OuterLinear_3702261809263
// MI455X (gfx1250) — matrix operand order repaired and checked by reading, not compiled, not run on hardware
//
#include <hip/hip_runtime.h>

typedef __attribute__((ext_vector_type(16))) _Float16 v16h;
typedef __attribute__((ext_vector_type(8)))  _Float16 v8h;
typedef __attribute__((ext_vector_type(4)))  _Float16 v4h;
typedef __attribute__((ext_vector_type(8)))  float    v8f;
typedef __attribute__((ext_vector_type(4)))  float    v4f;
typedef __attribute__((ext_vector_type(4)))  unsigned int v4u;

#define FEAT   1024
#define MTOT   8192      // 4 * 2048
#define NTOT   16384     // 32*32*16
#define BM     256
#define BN     128
#define BK     32
#define KTILES (FEAT / BK)   // 32
#define LDST   40            // LDS row stride in halves (32 + 8 pad, keeps 16B align)
#define EPS    1e-6f

// ---------------- utility kernels ----------------

__global__ __launch_bounds__(256) void zero_rowsq_kernel(float* __restrict__ rowsq) {
    int i = blockIdx.x * 256 + threadIdx.x;
    if (i < MTOT) rowsq[i] = 0.0f;
}

__global__ __launch_bounds__(256) void cvt_x_kernel(const float* __restrict__ x,
                                                    _Float16* __restrict__ xh) {
    size_t i = ((size_t)blockIdx.x * 256 + threadIdx.x) * 4;
    v4f v = *(const v4f*)(x + i);
    v4h h;
    h[0] = (_Float16)v[0]; h[1] = (_Float16)v[1];
    h[2] = (_Float16)v[2]; h[3] = (_Float16)v[3];
    *(v4h*)(xh + i) = h;
}

// W[o][z] = a[o>>9][z] * b[(o>>4)&31][z] * c[o&15][z], stored f16
__global__ __launch_bounds__(256) void build_w_kernel(const float* __restrict__ w,
                                                      _Float16* __restrict__ wh) {
    size_t idx = (size_t)blockIdx.x * 256 + threadIdx.x;
    int o = (int)(idx >> 8);          // 0..16383
    int z = ((int)idx & 255) * 4;     // 0..1020
    int ia = o >> 9, ib = (o >> 4) & 31, ic = o & 15;
    v4f va = *(const v4f*)(w + (size_t)ia        * FEAT + z);
    v4f vb = *(const v4f*)(w + (size_t)(32 + ib) * FEAT + z);
    v4f vc = *(const v4f*)(w + (size_t)(64 + ic) * FEAT + z);
    v4h h;
    h[0] = (_Float16)(va[0] * vb[0] * vc[0]);
    h[1] = (_Float16)(va[1] * vb[1] * vc[1]);
    h[2] = (_Float16)(va[2] * vb[2] * vc[2]);
    h[3] = (_Float16)(va[3] * vb[3] * vc[3]);
    *(v4h*)(wh + (size_t)o * FEAT + z) = h;
}

__global__ __launch_bounds__(256) void rmsnorm_kernel(float* __restrict__ y,
                                                      const float* __restrict__ rowsq) {
    size_t i = ((size_t)blockIdx.x * 256 + threadIdx.x) * 4;
    int row = (int)(i >> 14);   // NTOT = 16384 per row
    float s = rsqrtf(rowsq[row] * (1.0f / (float)NTOT) + EPS);
    v4f v = *(v4f*)(y + i);
    v[0] *= s; v[1] *= s; v[2] *= s; v[3] *= s;
    *(v4f*)(y + i) = v;
}

// ---------------- main GEMM: y = x_f16 * W_f16^T, + per-row sum(y^2) ----------------
// Block tile 256x128, K-step 32, 8 waves (4 M x 2 N), wave tile 64x64 (4x4 WMMA accs).

__global__ __launch_bounds__(256) void gemm_wmma_kernel(
    const _Float16* __restrict__ X,   // (MTOT, FEAT)
    const _Float16* __restrict__ W,   // (NTOT, FEAT)
    float* __restrict__ Y,            // (MTOT, NTOT)
    float* __restrict__ rowsq)        // (MTOT)
{
    __shared__ _Float16 lA[2][BM * LDST];   // 2 x 20 KB
    __shared__ _Float16 lB[2][BN * LDST];   // 2 x 10 KB

    const int tid  = threadIdx.x;
    const int lane = tid & 31;
    const int wave = tid >> 5;
    const int wr   = wave & 3;    // M quadrant (64 rows each)
    const int wc   = wave >> 2;   // N half (64 cols each)

    const int bm = blockIdx.y * BM;
    const int bn = blockIdx.x * BN;

    // Cooperative tile load. A tile: 256x32 halves = 1024 x 16B chunks (4/thread).
    // B tile: 128x32 halves = 512 chunks (2/thread).
    const int lrow = tid >> 2;            // 0..63
    const int lcol = (tid & 3) * 8;       // 0,8,16,24 (halves)
    const int offS = lrow * LDST + lcol;  // LDS store offset (halves)

    const _Float16* gA = X + (size_t)(bm + lrow) * FEAT + lcol;
    const _Float16* gB = W + (size_t)(bn + lrow) * FEAT + lcol;

    // Prologue: tile 0 -> LDS buffer 0
    {
        v4u a0 = *(const v4u*)(gA);
        v4u a1 = *(const v4u*)(gA + (size_t)64  * FEAT);
        v4u a2 = *(const v4u*)(gA + (size_t)128 * FEAT);
        v4u a3 = *(const v4u*)(gA + (size_t)192 * FEAT);
        v4u b0 = *(const v4u*)(gB);
        v4u b1 = *(const v4u*)(gB + (size_t)64  * FEAT);
        *(v4u*)&lA[0][offS]              = a0;
        *(v4u*)&lA[0][offS +  64 * LDST] = a1;
        *(v4u*)&lA[0][offS + 128 * LDST] = a2;
        *(v4u*)&lA[0][offS + 192 * LDST] = a3;
        *(v4u*)&lB[0][offS]              = b0;
        *(v4u*)&lB[0][offS +  64 * LDST] = b1;
    }

    // Fragment addressing (ISA VGPR layouts, wave32)
    const int fr    = lane & 15;           // row within 16x16 tile
    const int kA    = (lane >> 4) * 8;     // A: halves 0..7 -> K kA+0..7; 8..15 -> K 16+kA+0..7
    const int kB    = (lane >> 4) * 8;     // B: halves 0..7 -> K kB+0..7; 8..15 -> K 16+kB+0..7
    const int aoff0 = (64 * wr + fr) * LDST;
    const int boff0 = (64 * wc + fr) * LDST;

    v8f acc[4][4];
    #pragma unroll
    for (int mi = 0; mi < 4; ++mi)
        #pragma unroll
        for (int ni = 0; ni < 4; ++ni)
            #pragma unroll
            for (int e = 0; e < 8; ++e) acc[mi][ni][e] = 0.0f;

    union Frag { v16h v; v8h h[2]; };

    for (int kt = 0; kt < KTILES; ++kt) {
        __syncthreads();
        const int cur  = kt & 1;
        const bool more = (kt + 1) < KTILES;

        // Issue next tile's global loads early (hide L2 latency under WMMA)
        v4u na0, na1, na2, na3, nb0, nb1;
        if (more) {
            const size_t go = (size_t)(kt + 1) * BK;
            na0 = *(const v4u*)(gA + go);
            na1 = *(const v4u*)(gA + go + (size_t)64  * FEAT);
            na2 = *(const v4u*)(gA + go + (size_t)128 * FEAT);
            na3 = *(const v4u*)(gA + go + (size_t)192 * FEAT);
            nb0 = *(const v4u*)(gB + go);
            nb1 = *(const v4u*)(gB + go + (size_t)64  * FEAT);
        }

        Frag aF[4], bF[4];
        #pragma unroll
        for (int mi = 0; mi < 4; ++mi) {
            const int base = aoff0 + mi * 16 * LDST;
            aF[mi].h[0] = *(const v8h*)&lA[cur][base + kA];
            aF[mi].h[1] = *(const v8h*)&lA[cur][base + 16 + kA];
        }
        #pragma unroll
        for (int ni = 0; ni < 4; ++ni) {
            const int base = boff0 + ni * 16 * LDST;
            bF[ni].h[0] = *(const v8h*)&lB[cur][base + kB];
            bF[ni].h[1] = *(const v8h*)&lB[cur][base + 16 + kB];
        }

        #pragma unroll
        for (int mi = 0; mi < 4; ++mi)
            #pragma unroll
            for (int ni = 0; ni < 4; ++ni)
                acc[mi][ni] = __builtin_amdgcn_wmma_f32_16x16x32_f16(
                    false, aF[mi].v, false, bF[ni].v,
                    (short)0, acc[mi][ni], false, false);

        if (more) {
            const int nxt = cur ^ 1;
            *(v4u*)&lA[nxt][offS]              = na0;
            *(v4u*)&lA[nxt][offS +  64 * LDST] = na1;
            *(v4u*)&lA[nxt][offS + 128 * LDST] = na2;
            *(v4u*)&lA[nxt][offS + 192 * LDST] = na3;
            *(v4u*)&lB[nxt][offS]              = nb0;
            *(v4u*)&lB[nxt][offS +  64 * LDST] = nb1;
        }
    }

    // Epilogue: write y, accumulate per-row sum of squares.
    // C/D layout: element v of accumulator -> row M = v + 8*(lane>>4), col N = lane&15.
    const int rbase = bm + 64 * wr + 8 * (lane >> 4);
    const int cbase = bn + 64 * wc + (lane & 15);

    #pragma unroll
    for (int mi = 0; mi < 4; ++mi) {
        float ss[8];
        #pragma unroll
        for (int v = 0; v < 8; ++v) ss[v] = 0.0f;

        #pragma unroll
        for (int ni = 0; ni < 4; ++ni) {
            const int col = cbase + 16 * ni;
            #pragma unroll
            for (int v = 0; v < 8; ++v) {
                const float t = acc[mi][ni][v];
                Y[(size_t)(rbase + 16 * mi + v) * NTOT + col] = t;
                ss[v] += t * t;
            }
        }
        // reduce across the 16 lanes sharing the same rows (xor masks stay within half)
        #pragma unroll
        for (int m = 1; m < 16; m <<= 1)
            #pragma unroll
            for (int v = 0; v < 8; ++v) ss[v] += __shfl_xor(ss[v], m, 32);

        if ((lane & 15) == 0) {
            #pragma unroll
            for (int v = 0; v < 8; ++v)
                atomicAdd(&rowsq[rbase + 16 * mi + v], ss[v]);
        }
    }
}

// ---------------- launcher ----------------

extern "C" void kernel_launch(void* const* d_in, const int* in_sizes, int n_in,
                              void* d_out, int out_size, void* d_ws, size_t ws_size,
                              hipStream_t stream) {
    const float* x = (const float*)d_in[0];   // (4,2048,1024) f32
    const float* w = (const float*)d_in[1];   // (80,1024) f32
    float* y = (float*)d_out;                 // (4,2048,16384) f32

    _Float16* xh    = (_Float16*)d_ws;                         // 16 MB
    _Float16* wh    = xh + (size_t)MTOT * FEAT;                // 32 MB
    float*    rowsq = (float*)(wh + (size_t)NTOT * FEAT);      // 32 KB

    zero_rowsq_kernel<<<dim3(MTOT / 256), dim3(256), 0, stream>>>(rowsq);
    cvt_x_kernel<<<dim3((MTOT * FEAT) / (256 * 4)), dim3(256), 0, stream>>>(x, xh);
    build_w_kernel<<<dim3(NTOT), dim3(256), 0, stream>>>(w, wh);

    gemm_wmma_kernel<<<dim3(NTOT / BN, MTOT / BM), dim3(256), 0, stream>>>(xh, wh, y, rowsq);

    rmsnorm_kernel<<<dim3((unsigned)(((size_t)MTOT * NTOT) / (256 * 4))), dim3(256), 0, stream>>>(y, rowsq);
}
